// WaveletCBAM_32710470926739
// MI455X (gfx1250) — matrix operand order repaired and checked by reading, not compiled, not run on hardware
//
#include <hip/hip_runtime.h>
#include <hip/hip_bf16.h>
#include <math.h>

typedef __attribute__((ext_vector_type(16))) _Float16 v16h;
typedef __attribute__((ext_vector_type(8)))  float    v8f;

#define B_    8
#define C_    128
#define H_    256
#define W_    256
#define HO_   129
#define WO_   129
#define HID_  11
#define PLANE_ (HO_*WO_)          // 16641

// fr = filt[::-1]
__device__ __constant__ float FR_LO[4] = { 0.48296291314453416f, 0.8365163037378079f,
                                           0.2241438680420134f, -0.12940952255126037f };
__device__ __constant__ float FR_HI[4] = { -0.12940952255126037f, -0.2241438680420134f,
                                           0.8365163037378079f, -0.48296291314453416f };

// symmetric ('symmetric') boundary reflection, pad=3 < L so one reflection suffices
__device__ __forceinline__ int symref(int o, int L) {
    if (o < 0)   o = -o - 1;
    if (o >= L)  o = 2 * L - 1 - o;
    return o;
}

// monotone float <-> sortable uint mapping (for atomicMax on signed floats)
__device__ __forceinline__ unsigned fmap(float f) {
    unsigned u = __float_as_uint(f);
    return (u & 0x80000000u) ? ~u : (u | 0x80000000u);
}
__device__ __forceinline__ float funmap(unsigned u) {
    return (u & 0x80000000u) ? __uint_as_float(u & 0x7FFFFFFFu)
                             : __uint_as_float(~u);
}

// ---------------------------------------------------------------------------
// Kernel 1: fused DWT (db2, cA & cH) + all four reductions.
// grid.x = B*C (one (b,c) plane), grid.y tiles the 129x129 output plane.
// ---------------------------------------------------------------------------
__global__ void dwt_reduce_kernel(const float* __restrict__ x,
                                  float* __restrict__ chan_sum,     // [B*C] sum of cA
                                  unsigned* __restrict__ chan_max,  // [B*C] mapped max of cH
                                  float* __restrict__ sp_sum,       // [B*PLANE] sum_c (cA+cH)
                                  unsigned* __restrict__ sp_max) {  // [B*PLANE] mapped max_c max(cA,cH)
    const int bc = blockIdx.x;
    const int b  = bc / C_;
    const int idx = blockIdx.y * blockDim.x + threadIdx.x;

    float a_val = 0.0f;
    float h_val = -INFINITY;

    if (idx < PLANE_) {
        const int i = idx / WO_;
        const int j = idx % WO_;
        const float* xp = x + (size_t)bc * H_ * W_;

        // column indices for the W-direction lowpass (shared across the 4 rows)
        int col[4];
        #pragma unroll
        for (int k = 0; k < 4; ++k) col[k] = symref(2 * j + k - 2, W_);

        float low[4];
        #pragma unroll
        for (int m = 0; m < 4; ++m) {
            const int r = symref(2 * i + m - 2, H_);
            const float* rowp = xp + (size_t)r * W_;
            float s = 0.0f;
            #pragma unroll
            for (int k = 0; k < 4; ++k) s = fmaf(FR_LO[k], rowp[col[k]], s);
            low[m] = s;
        }
        float a = 0.0f, hh = 0.0f;
        #pragma unroll
        for (int m = 0; m < 4; ++m) {
            a  = fmaf(FR_LO[m], low[m], a);
            hh = fmaf(FR_HI[m], low[m], hh);
        }

        // per-pixel channel reductions (mean over 2C channels, max over 2C channels)
        atomicAdd(&sp_sum[b * PLANE_ + idx], a + hh);
        atomicMax(&sp_max[b * PLANE_ + idx], fmap(fmaxf(a, hh)));

        a_val = a;
        h_val = hh;
    }

    // block tree reduction for per-(b,c) stats
    __shared__ float ssum[256];
    __shared__ float smax[256];
    ssum[threadIdx.x] = a_val;
    smax[threadIdx.x] = h_val;
    __syncthreads();
    for (int s = 128; s > 0; s >>= 1) {
        if ((int)threadIdx.x < s) {
            ssum[threadIdx.x] += ssum[threadIdx.x + s];
            smax[threadIdx.x]  = fmaxf(smax[threadIdx.x], smax[threadIdx.x + s]);
        }
        __syncthreads();
    }
    if (threadIdx.x == 0) {
        atomicAdd(&chan_sum[bc], ssum[0]);
        atomicMax(&chan_max[bc], fmap(smax[0]));
    }
}

// ---------------------------------------------------------------------------
// Kernel 2: channel-attention MLP on one wave (wave32) with WMMA.
//   p = avg_pool(cA) + max_pool(cH)                      (8 x 128)
//   h = relu(p @ w1 + b1)                                (8 x 11)
//   att = sigmoid(h @ w2 + b2)                           (8 x 128)
// Uses V_WMMA_F32_16X16X32_F16, operands staged in LDS with the ISA lane layout.
// ---------------------------------------------------------------------------
__device__ __forceinline__ v16h load_a_frag(const _Float16* A, int lda, int kbase, int lane) {
    // 16-bit A 16x32 layout: lane<16 holds K 0..7 & 16..23; lane>=16 holds K 8..15 & 24..31
    const int m  = lane & 15;
    const int kh = (lane < 16) ? 0 : 8;
    v16h f;
    #pragma unroll
    for (int h = 0; h < 8; ++h) f[h]     = A[m * lda + kbase + kh + h];
    #pragma unroll
    for (int h = 0; h < 8; ++h) f[8 + h] = A[m * lda + kbase + 16 + kh + h];
    return f;
}
__device__ __forceinline__ v16h load_b_frag(const _Float16* Bm, int ldb, int kbase, int nbase, int lane) {
    // B 32x16: same K order as A: lanes 0-15 hold K 0..7 & 16..23, lanes 16-31 hold K 8..15 & 24..31
    const int n  = lane & 15;
    const int k0 = (lane < 16) ? 0 : 8;
    v16h f;
    #pragma unroll
    for (int h = 0; h < 16; ++h) f[h] = Bm[(size_t)(kbase + k0 + (h < 8 ? h : h + 8)) * ldb + nbase + n];
    return f;
}

__global__ void mlp_attention_kernel(const float* __restrict__ chan_sum,
                                     const unsigned* __restrict__ chan_max,
                                     const float* __restrict__ w1,
                                     const float* __restrict__ b1,
                                     const float* __restrict__ w2,
                                     const float* __restrict__ b2,
                                     float* __restrict__ att) {
    __shared__ _Float16 p16[16 * 128];   // A for step 1 (rows 8..15 zero)
    __shared__ _Float16 w1h[128 * 16];   // B for step 1 (cols 11..15 zero)
    __shared__ _Float16 h16[16 * 32];    // A for step 2 (K cols 11..31 zero)
    __shared__ _Float16 w2h[32 * 128];   // B for step 2 (K rows 11..31 zero)

    const int lane = threadIdx.x;        // 32 threads = one wave

    const float inv_plane = 1.0f / (float)PLANE_;
    for (int i = lane; i < 16 * 128; i += 32) {
        const int m = i / 128, k = i % 128;
        float v = 0.0f;
        if (m < B_) v = chan_sum[m * C_ + k] * inv_plane + funmap(chan_max[m * C_ + k]);
        p16[i] = (_Float16)v;
    }
    for (int i = lane; i < 128 * 16; i += 32) {
        const int n = i % 16;
        const int k = i / 16;
        w1h[i] = (_Float16)((n < HID_) ? w1[k * HID_ + n] : 0.0f);
    }
    for (int i = lane; i < 16 * 32; i += 32) h16[i] = (_Float16)0.0f;
    for (int i = lane; i < 32 * 128; i += 32) {
        const int k = i / 128, n = i % 128;
        w2h[i] = (_Float16)((k < HID_) ? w2[k * 128 + n] : 0.0f);
    }
    __syncthreads();

    const int n     = lane & 15;
    const int mbase = (lane < 16) ? 0 : 8;

    // ---- step 1: h = relu(p @ w1 + b1), K = 128 in 4 chunks of 32 ----
    v8f acc = {};
    #pragma unroll
    for (int kb = 0; kb < 128; kb += 32) {
        v16h a  = load_a_frag(p16, 128, kb, lane);
        v16h bb = load_b_frag(w1h, 16, kb, 0, lane);
        acc = __builtin_amdgcn_wmma_f32_16x16x32_f16(false, a, false, bb,
                                                     (short)0, acc, false, false);
    }
    const float bias1 = (n < HID_) ? b1[n] : 0.0f;
    #pragma unroll
    for (int r = 0; r < 8; ++r) {
        const int m = mbase + r;
        float v = (n < HID_ && m < B_) ? (acc[r] + bias1) : 0.0f;
        v = v > 0.0f ? v : 0.0f;
        h16[m * 32 + n] = (_Float16)v;    // K cols 16..31 stay zero
    }
    __syncthreads();

    // ---- step 2: att = sigmoid(h @ w2 + b2), N = 128 in 8 tiles of 16 ----
    const v16h a2 = load_a_frag(h16, 32, 0, lane);
    #pragma unroll
    for (int nb = 0; nb < 128; nb += 16) {
        v16h bb = load_b_frag(w2h, 128, 0, nb, lane);
        v8f acc2 = {};
        acc2 = __builtin_amdgcn_wmma_f32_16x16x32_f16(false, a2, false, bb,
                                                      (short)0, acc2, false, false);
        const int col = nb + n;
        const float bias2 = b2[col];
        #pragma unroll
        for (int r = 0; r < 8; ++r) {
            const int m = mbase + r;
            if (m < B_) {
                const float v = acc2[r] + bias2;
                att[m * C_ + col] = 1.0f / (1.0f + expf(-v));
            }
        }
    }
}

// ---------------------------------------------------------------------------
// Kernel 3: spatial attention: 5x5 conv over (channel-mean, channel-max) + sigmoid.
// ---------------------------------------------------------------------------
__global__ void spatial_att_kernel(const float* __restrict__ sp_sum,
                                   const unsigned* __restrict__ sp_max,
                                   const float* __restrict__ w_sa,
                                   float* __restrict__ sa) {
    const int idx = blockIdx.x * blockDim.x + threadIdx.x;
    if (idx >= B_ * PLANE_) return;
    const int b   = idx / PLANE_;
    const int rem = idx % PLANE_;
    const int i   = rem / WO_;
    const int j   = rem % WO_;
    const float inv2C = 1.0f / (float)(2 * C_);

    float acc = 0.0f;
    #pragma unroll
    for (int di = 0; di < 5; ++di) {
        const int ii = i + di - 2;
        if (ii < 0 || ii >= HO_) continue;
        #pragma unroll
        for (int dj = 0; dj < 5; ++dj) {
            const int jj = j + dj - 2;
            if (jj < 0 || jj >= WO_) continue;
            const int p = b * PLANE_ + ii * WO_ + jj;
            const float avg = sp_sum[p] * inv2C;
            const float mx  = funmap(sp_max[p]);
            acc = fmaf(w_sa[di * 5 + dj],      avg, acc);
            acc = fmaf(w_sa[25 + di * 5 + dj], mx,  acc);
        }
    }
    sa[idx] = 1.0f / (1.0f + expf(-acc));
}

// ---------------------------------------------------------------------------
// Kernel 4: fused gating: out = x * channel_att[b,c] * bilinear(sa)[b,h,w].
// Streaming, bandwidth-bound: 256 MB read + 256 MB write.
// ---------------------------------------------------------------------------
__global__ void gate_kernel(const float* __restrict__ x,
                            const float* __restrict__ att,
                            const float* __restrict__ sa,
                            float* __restrict__ out) {
    const size_t idx = (size_t)blockIdx.x * blockDim.x + threadIdx.x;
    const size_t total = (size_t)B_ * C_ * H_ * W_;
    if (idx >= total) return;

    __builtin_prefetch(x + idx + 4096, 0, 1);   // global_prefetch_b8

    const int w = (int)(idx % W_);
    size_t t = idx / W_;
    const int h = (int)(t % H_); t /= H_;
    const int c = (int)(t % C_);
    const int b = (int)(t / C_);

    // bilinear sample: src = (dst + 0.5) * (in/out) - 0.5, edge-clamped
    float sy = (h + 0.5f) * ((float)HO_ / (float)H_) - 0.5f;
    float sx = (w + 0.5f) * ((float)WO_ / (float)W_) - 0.5f;
    sy = fminf(fmaxf(sy, 0.0f), (float)(HO_ - 1));
    sx = fminf(fmaxf(sx, 0.0f), (float)(WO_ - 1));
    const int y0 = (int)sy;
    const int x0 = (int)sx;
    const int y1 = min(y0 + 1, HO_ - 1);
    const int x1 = min(x0 + 1, WO_ - 1);
    const float fy = sy - (float)y0;
    const float fx = sx - (float)x0;

    const float* sap = sa + (size_t)b * PLANE_;
    const float s00 = sap[y0 * WO_ + x0];
    const float s01 = sap[y0 * WO_ + x1];
    const float s10 = sap[y1 * WO_ + x0];
    const float s11 = sap[y1 * WO_ + x1];
    const float s0 = s00 + (s01 - s00) * fx;
    const float s1 = s10 + (s11 - s10) * fx;
    const float s  = s0 + (s1 - s0) * fy;

    out[idx] = x[idx] * att[b * C_ + c] * s;
}

// ---------------------------------------------------------------------------
extern "C" void kernel_launch(void* const* d_in, const int* in_sizes, int n_in,
                              void* d_out, int out_size, void* d_ws, size_t ws_size,
                              hipStream_t stream) {
    const float* x    = (const float*)d_in[0];
    const float* w1   = (const float*)d_in[1];
    const float* b1   = (const float*)d_in[2];
    const float* w2   = (const float*)d_in[3];
    const float* b2   = (const float*)d_in[4];
    const float* w_sa = (const float*)d_in[5];
    float* out = (float*)d_out;

    // workspace carve (all tiny; ~1.6 MB total)
    char* ws = (char*)d_ws;
    size_t off = 0;
    auto carve = [&](size_t bytes) -> void* {
        void* p = ws + off;
        off = (off + bytes + 255) & ~(size_t)255;
        return p;
    };
    float*    chan_sum = (float*)   carve((size_t)B_ * C_ * sizeof(float));
    unsigned* chan_max = (unsigned*)carve((size_t)B_ * C_ * sizeof(unsigned));
    float*    sp_sum   = (float*)   carve((size_t)B_ * PLANE_ * sizeof(float));
    unsigned* sp_max   = (unsigned*)carve((size_t)B_ * PLANE_ * sizeof(unsigned));
    const size_t accum_bytes = off;   // everything above is an accumulator -> zero init
    float*    sa       = (float*)   carve((size_t)B_ * PLANE_ * sizeof(float));
    float*    att      = (float*)   carve((size_t)B_ * C_ * sizeof(float));

    // zero accumulators (0u is the identity for the sortable-uint max encoding too)
    hipMemsetAsync(d_ws, 0, accum_bytes, stream);

    // 1) fused DWT + reductions
    {
        dim3 grid(B_ * C_, (PLANE_ + 255) / 256);
        dwt_reduce_kernel<<<grid, 256, 0, stream>>>(x, chan_sum, chan_max, sp_sum, sp_max);
    }
    // 2) channel-attention MLP (one wave, WMMA)
    mlp_attention_kernel<<<1, 32, 0, stream>>>(chan_sum, chan_max, w1, b1, w2, b2, att);
    // 3) spatial attention conv + sigmoid
    {
        const int n = B_ * PLANE_;
        spatial_att_kernel<<<(n + 255) / 256, 256, 0, stream>>>(sp_sum, sp_max, w_sa, sa);
    }
    // 4) fused bilinear-upsample + gating
    {
        const size_t total = (size_t)B_ * C_ * H_ * W_;
        gate_kernel<<<(unsigned)((total + 255) / 256), 256, 0, stream>>>(x, att, sa, out);
    }
}
